// BitFeedForward_39393440039203
// MI455X (gfx1250) — 8-bit matrix operand order repaired and checked by reading, compiled, not run on hardware
//
#include <hip/hip_runtime.h>
#include <hip/hip_bf16.h>
#include <stdint.h>


#define USE_ASYNC_COPY 1

typedef int v8i __attribute__((ext_vector_type(8)));
typedef uint64_t u64a __attribute__((may_alias, aligned(8)));

union Frag {
    v8i v;
    uint64_t q[4];
};


__global__ void zero_sums_kernel(float* s) {
    if (threadIdx.x < 2) s[threadIdx.x] = 0.0f;
}

__global__ __launch_bounds__(256) void absmean_reduce_kernel(
    const float* __restrict__ w, int n, float* __restrict__ out) {
    __shared__ float red[256];
    float s = 0.0f;
    for (int i = blockIdx.x * 256 + threadIdx.x; i < n; i += gridDim.x * 256)
        s += fabsf(w[i]);
    red[threadIdx.x] = s;
    __syncthreads();
    for (int k = 128; k > 0; k >>= 1) {
        if ((int)threadIdx.x < k) red[threadIdx.x] += red[threadIdx.x + k];
        __syncthreads();
    }
    if (threadIdx.x == 0) atomicAdd(out, red[0]);
}

__global__ __launch_bounds__(256) void wquant_kernel(
    const float* __restrict__ w, int n, const float* __restrict__ sum,
    int8_t* __restrict__ q) {
    const int i = blockIdx.x * 256 + threadIdx.x;
    if (i >= n) return;
    const float mean = fmaxf(*sum / (float)n, 1e-5f);
    float t = rintf(w[i] / mean);
    t = fminf(fmaxf(t, -1.0f), 1.0f);
    q[i] = (int8_t)t;
}


template <int D, int E>
__device__ __forceinline__ void act_quant_core(float (&v)[E], size_t row,
                                               int8_t* __restrict__ q,
                                               float* __restrict__ inv_scale,
                                               float ss) {
    __shared__ float red[256];
    const int tid = threadIdx.x;
    red[tid] = ss;
    __syncthreads();
    for (int s = 128; s > 0; s >>= 1) {
        if (tid < s) red[tid] += red[tid + s];
        __syncthreads();
    }
    const float rms = rsqrtf(red[0] / (float)D + 1e-5f);
    __syncthreads();

    float amax = 0.0f;
#pragma unroll
    for (int e = 0; e < E; ++e) {
        v[e] *= rms;
        amax = fmaxf(amax, fabsf(v[e]));
    }
    red[tid] = amax;
    __syncthreads();
    for (int s = 128; s > 0; s >>= 1) {
        if (tid < s) red[tid] = fmaxf(red[tid], red[tid + s]);
        __syncthreads();
    }
    const float mx = fmaxf(red[0], 1e-5f);
    const float scale = 127.0f / mx;
    if (tid == 0) inv_scale[row] = mx / 127.0f;

    signed char o[E];
#pragma unroll
    for (int e = 0; e < E; ++e) {
        float t = rintf(v[e] * scale);
        t = fminf(fmaxf(t, -128.0f), 127.0f);
        o[e] = (signed char)t;
    }
    int8_t* qr = q + row * (size_t)D + (size_t)tid * E;
#pragma unroll
    for (int c = 0; c < E / 4; ++c)
        *(char4*)(qr + 4 * c) =
            make_char4(o[4 * c], o[4 * c + 1], o[4 * c + 2], o[4 * c + 3]);
}

template <int D>
__global__ __launch_bounds__(256) void act_quant_f32_kernel(
    const float* __restrict__ x, int8_t* __restrict__ q,
    float* __restrict__ inv_scale) {
    constexpr int E = D / 256;
    const size_t row = blockIdx.x;
    const float* xr = x + row * (size_t)D + (size_t)threadIdx.x * E;
    float v[E];
    float ss = 0.0f;
#pragma unroll
    for (int e = 0; e < E; ++e) {
        v[e] = xr[e];
        ss += v[e] * v[e];
    }
    act_quant_core<D, E>(v, row, q, inv_scale, ss);
}

template <int D>
__global__ __launch_bounds__(256) void act_quant_bf16_kernel(
    const uint32_t* __restrict__ x, int8_t* __restrict__ q,
    float* __restrict__ inv_scale) {
    constexpr int E = D / 256;
    const size_t row = blockIdx.x;
    const uint32_t* xr = x + row * (size_t)(D / 2) + (size_t)threadIdx.x * (E / 2);
    float v[E];
    float ss = 0.0f;
#pragma unroll
    for (int c = 0; c < E / 2; ++c) {
        const uint32_t u = xr[c];
        v[2 * c + 0] = __uint_as_float(u << 16);
        v[2 * c + 1] = __uint_as_float(u & 0xffff0000u);
        ss += v[2 * c] * v[2 * c] + v[2 * c + 1] * v[2 * c + 1];
    }
    act_quant_core<D, E>(v, row, q, inv_scale, ss);
}


constexpr int LDSS = 80;

__device__ __forceinline__ void async_b128(unsigned ldsoff, const int8_t* g) {
#if USE_ASYNC_COPY
    asm volatile("global_load_async_to_lds_b128 %0, %1, off"
                 :
                 : "v"(ldsoff), "v"(g)
                 : "memory");
#endif
}

__device__ __forceinline__ void wait_async_le(int n) {
#if USE_ASYNC_COPY
    if (n)
        asm volatile("s_wait_asynccnt %0" ::"i"(4) : "memory");
    else
        asm volatile("s_wait_asynccnt %0" ::"i"(0) : "memory");
#endif
}

template <int K, bool GELU, typename OutT>
__global__ __launch_bounds__(256) void bitgemm_wmma_iu8(
    const int8_t* __restrict__ A, const int8_t* __restrict__ B,
    const float* __restrict__ ainv, const float* __restrict__ wsum,
    float wcount, const float* __restrict__ bias, OutT* __restrict__ out,
    int N) {
    __shared__ __align__(16) int8_t lA[2][128 * LDSS];
    __shared__ __align__(16) int8_t lB[2][128 * LDSS];

    const int tid = threadIdx.x;
    const int lane = tid & 31;
    const int wave = tid >> 5;
    const int l15 = lane & 15;
    const int hi = lane >> 4;
    const int wm = wave & 3;
    const int wn = wave >> 2;
    const int bm = blockIdx.y * 128;
    const int bn = blockIdx.x * 128;

    const int crow = tid >> 1;
    const int chalf = (tid & 1) * 32;
    const int8_t* gA = A + (size_t)(bm + crow) * K + chalf;
    const int8_t* gB = B + (size_t)(bn + crow) * K + chalf;
    const unsigned lrow = (unsigned)(crow * LDSS + chalf);
    const unsigned ldsA[2] = {(unsigned)(uintptr_t)&lA[0][0] + lrow,
                              (unsigned)(uintptr_t)&lA[1][0] + lrow};
    const unsigned ldsB[2] = {(unsigned)(uintptr_t)&lB[0][0] + lrow,
                              (unsigned)(uintptr_t)&lB[1][0] + lrow};

#if !USE_ASYNC_COPY
    int8_t* sA[2] = {&lA[0][lrow - 0], &lA[1][lrow - 0]};
    int8_t* sB[2] = {&lB[0][lrow - 0], &lB[1][lrow - 0]};
#endif

    auto issue = [&](int b, int k0) {
#if USE_ASYNC_COPY
        async_b128(ldsA[b] + 0, gA + k0 + 0);
        async_b128(ldsA[b] + 16, gA + k0 + 16);
        async_b128(ldsB[b] + 0, gB + k0 + 0);
        async_b128(ldsB[b] + 16, gB + k0 + 16);
#else
        *(int4*)(sA[b] + 0) = *(const int4*)(gA + k0 + 0);
        *(int4*)(sA[b] + 16) = *(const int4*)(gA + k0 + 16);
        *(int4*)(sB[b] + 0) = *(const int4*)(gB + k0 + 0);
        *(int4*)(sB[b] + 16) = *(const int4*)(gB + k0 + 16);
#endif
    };

    const int aoff0 = (wm * 32 + l15) * LDSS + hi * 16;
    const int aoff1 = aoff0 + 16 * LDSS;
    int boff[4];
#pragma unroll
    for (int j = 0; j < 4; ++j)
        boff[j] = (wn * 64 + j * 16 + l15) * LDSS + hi * 16;

    const v8i vzero = {0, 0, 0, 0, 0, 0, 0, 0};
    v8i acc[2][4];
#pragma unroll
    for (int i = 0; i < 2; ++i)
#pragma unroll
        for (int j = 0; j < 4; ++j) acc[i][j] = vzero;

    constexpr int NS = K / 64;
    issue(0, 0);
    for (int s = 0; s < NS; ++s) {
        const int buf = s & 1;
        if (s + 1 < NS) {
            issue(buf ^ 1, (s + 1) * 64);
            wait_async_le(4);
        } else {
            wait_async_le(0);
        }
        __syncthreads();

        const int8_t* tA = &lA[buf][0];
        const int8_t* tB = &lB[buf][0];
        Frag a[2], b[4];
#pragma unroll
        for (int c = 0; c < 4; ++c) {
            a[0].q[c] = *(const u64a*)(tA + aoff0 + (c & 1) * 8 + (c >> 1) * 32);
            a[1].q[c] = *(const u64a*)(tA + aoff1 + (c & 1) * 8 + (c >> 1) * 32);
        }
#pragma unroll
        for (int j = 0; j < 4; ++j) {
            b[j].q[0] = *(const u64a*)(tB + boff[j] + 0);
            b[j].q[1] = *(const u64a*)(tB + boff[j] + 8);
            b[j].q[2] = *(const u64a*)(tB + boff[j] + 32);
            b[j].q[3] = *(const u64a*)(tB + boff[j] + 40);
        }
#pragma unroll
        for (int i = 0; i < 2; ++i)
#pragma unroll
            for (int j = 0; j < 4; ++j)
                acc[i][j] = __builtin_amdgcn_wmma_i32_16x16x64_iu8(
                    true, a[i].v, true, b[j].v, acc[i][j], false, false);

        __syncthreads();
    }

    const float wsc = fmaxf(*wsum / wcount, 1e-5f);
#pragma unroll
    for (int i = 0; i < 2; ++i) {
#pragma unroll
        for (int r = 0; r < 8; ++r) {
            const int m = bm + wm * 32 + i * 16 + hi * 8 + r;
            const float as = ainv[m] * wsc;
#pragma unroll
            for (int j = 0; j < 4; ++j) {
                const int n = bn + wn * 64 + j * 16 + l15;
                float v = (float)acc[i][j][r] * as + bias[n];
                if (GELU) v = 0.5f * v * (1.0f + erff(v * 0.70710678118654752f));
                out[(size_t)m * N + n] = (OutT)v;
            }
        }
    }
}


extern "C" void kernel_launch(void* const* d_in, const int* in_sizes, int n_in,
                              void* d_out, int out_size, void* d_ws, size_t ws_size,
                              hipStream_t stream) {
    (void)in_sizes; (void)n_in; (void)out_size; (void)ws_size;
    constexpr int M = 2 * 8192;
    constexpr int D = 1024;
    constexpr int F = 4096;

    const float* x  = (const float*)d_in[0];
    const float* w1 = (const float*)d_in[1];
    const float* b1 = (const float*)d_in[2];
    const float* w2 = (const float*)d_in[3];
    const float* b2 = (const float*)d_in[4];
    float* out = (float*)d_out;

    char* ws = (char*)d_ws;
    size_t off = 0;
    auto take = [&](size_t bytes) -> char* {
        char* p = ws + off;
        off = (off + bytes + 255) & ~(size_t)255;
        return p;
    };
    float*          sums  = (float*)take(2 * sizeof(float));
    float*          a1inv = (float*)take((size_t)M * sizeof(float));
    float*          a2inv = (float*)take((size_t)M * sizeof(float));
    int8_t*         w1q   = (int8_t*)take((size_t)F * D);
    int8_t*         w2q   = (int8_t*)take((size_t)D * F);
    int8_t*         xq    = (int8_t*)take((size_t)M * D);
    int8_t*         hq    = (int8_t*)take((size_t)M * F);
    __hip_bfloat16* h     = (__hip_bfloat16*)take((size_t)M * F * sizeof(__hip_bfloat16));

    zero_sums_kernel<<<1, 32, 0, stream>>>(sums);
    absmean_reduce_kernel<<<1024, 256, 0, stream>>>(w1, F * D, sums + 0);
    absmean_reduce_kernel<<<1024, 256, 0, stream>>>(w2, D * F, sums + 1);
    wquant_kernel<<<(F * D) / 256, 256, 0, stream>>>(w1, F * D, sums + 0, w1q);
    wquant_kernel<<<(D * F) / 256, 256, 0, stream>>>(w2, D * F, sums + 1, w2q);

    act_quant_f32_kernel<D><<<M, 256, 0, stream>>>(x, xq, a1inv);
    bitgemm_wmma_iu8<D, true, __hip_bfloat16>
        <<<dim3(F / 128, M / 128), 256, 0, stream>>>(
            xq, w1q, a1inv, sums + 0, (float)(F * D), b1, h, F);

    act_quant_bf16_kernel<F><<<M, 256, 0, stream>>>(
        (const uint32_t*)h, hq, a2inv);
    bitgemm_wmma_iu8<F, false, float>
        <<<dim3(D / 128, M / 128), 256, 0, stream>>>(
            hq, w2q, a2inv, sums + 1, (float)(D * F), b2, out, D);
}
